// DecouplingLearningtoRotateAttentionLayer_6983616823304
// MI455X (gfx1250) — matrix operand order repaired and checked by reading, not compiled, not run on hardware
//
#include <hip/hip_runtime.h>
#include <math.h>

// ---------------- problem constants ----------------
#define Bc   8
#define Lc   3072
#define Sc   3072
#define Dc   512
#define Hc   8
#define Mc   2
#define Ec   64
#define INDc 96
#define PI_F 3.14159265358979323846f

typedef __attribute__((ext_vector_type(16))) __bf16 bf16x16;
typedef __attribute__((ext_vector_type(8)))  __bf16 bf16x8;
typedef __attribute__((ext_vector_type(4)))  __bf16 bf16x4;
typedef __attribute__((ext_vector_type(8)))  float  floatx8;
typedef __attribute__((ext_vector_type(4)))  float  floatx4;

static __device__ inline bf16x4 cvt4(floatx4 v)
{
  bf16x4 o;
  o[0] = (__bf16)v[0]; o[1] = (__bf16)v[1];
  o[2] = (__bf16)v[2]; o[3] = (__bf16)v[3];
  return o;
}

// =====================================================================
// Generic batched GEMM, C = act(alpha * A@B + bias), bf16 WMMA, f32 accum
//   A: [M,K] row-major (per (b,h) base = A + b*aB + h*aH), leading dim lda
//   B: if transB==0: [K,N] ldb ; if transB==1: [N,K] ldb
//   C: [M,N] ldc ; act: 0 none, 1 relu, 2 tanh(x)*pi
// Block tile 128x64xK32; 256 thr = 8 wave32; each wave computes 32x32
// (2x2 WMMA 16x16x32 tiles). Software-pipelined global->reg->LDS staging,
// branch-free via index clamping (OOB rows/cols are discarded in epilogue;
// K is always a multiple of 32 at every call site).
// All A/B base pointers + leading dims at every call site are 16B aligned.
// =====================================================================
#define BM 128
#define BN 64
#define BKK 32
#define LDS_K (BKK + 8)   // 40 bf16 = 80 B row stride (16B multiple)

__global__ __launch_bounds__(256)
void wmma_gemm_bf16(const float* __restrict__ A, const float* __restrict__ Bm,
                    const float* __restrict__ bias, float* __restrict__ C,
                    int Mrows, int Ncols, int K,
                    int lda, int ldb, int ldc,
                    long long aB, long long aH, long long bB, long long bH,
                    long long cB, long long cH,
                    int Hcnt, int transB, float alpha, int act)
{
  const int z  = blockIdx.z;
  const int bb = z / Hcnt;
  const int hh = z % Hcnt;
  A  += (size_t)bb * aB + (size_t)hh * aH;
  Bm += (size_t)bb * bB + (size_t)hh * bH;
  C  += (size_t)bb * cB + (size_t)hh * cH;

  __shared__ __bf16 As[BM][LDS_K];   // [m][k]
  __shared__ __bf16 Bs[BN][LDS_K];   // [n][k]  (transposed tile)

  const int tid  = threadIdx.x;
  const int lane = tid & 31;
  const int wave = tid >> 5;      // 0..7
  const int wm   = wave & 3;      // 4 waves along M (32 rows each)
  const int wn   = wave >> 2;     // 2 waves along N (32 cols each)
  const int half = lane >> 4;     // 0/1
  const int r    = lane & 15;

  const int m0 = blockIdx.x * BM;
  const int n0 = blockIdx.y * BN;

  // ---- staging coordinates (fixed per thread) ----
  int arow[4], acol[4];            // A tile: 128x32 = 4096 elems, 4 float4/thread
#pragma unroll
  for (int c = 0; c < 4; ++c) {
    const int chunk = tid + c * 256;      // 1024 chunks = 128 rows x 8 kgroups
    arow[c] = chunk >> 3;
    acol[c] = (chunk & 7) * 4;
  }
  int bn_[2], bk_[2];              // B tile: 64x32 = 2048 elems, 2 float4/thread
#pragma unroll
  for (int c = 0; c < 2; ++c) {
    const int chunk = tid + c * 256;      // 512 chunks
    if (transB) { bn_[c] = chunk >> 3; bk_[c] = (chunk & 7) * 4; } // contiguous k loads
    else        { bn_[c] = chunk & 63; bk_[c] = (chunk >> 6) * 4; } // coalesced n loads
  }

  floatx4 areg[4], breg[2];

  auto fetch = [&](int k0) {
#pragma unroll
    for (int c = 0; c < 4; ++c) {
      int gr = m0 + arow[c]; gr = gr < Mrows ? gr : Mrows - 1;   // clamp, not branch
      areg[c] = *(const floatx4*)(A + (size_t)gr * lda + (k0 + acol[c]));
    }
#pragma unroll
    for (int c = 0; c < 2; ++c) {
      int gn = n0 + bn_[c]; gn = gn < Ncols ? gn : Ncols - 1;
      if (transB) {
        breg[c] = *(const floatx4*)(Bm + (size_t)gn * ldb + (k0 + bk_[c]));
      } else {
        floatx4 v;
#pragma unroll
        for (int j = 0; j < 4; ++j)
          v[j] = Bm[(size_t)(k0 + bk_[c] + j) * ldb + gn];
        breg[c] = v;
      }
    }
  };

  auto stage = [&]() {
#pragma unroll
    for (int c = 0; c < 4; ++c)
      *(bf16x4*)&As[arow[c]][acol[c]] = cvt4(areg[c]);
#pragma unroll
    for (int c = 0; c < 2; ++c)
      *(bf16x4*)&Bs[bn_[c]][bk_[c]] = cvt4(breg[c]);
  };

  floatx8 acc[2][2];
#pragma unroll
  for (int tm = 0; tm < 2; ++tm)
#pragma unroll
    for (int tn = 0; tn < 2; ++tn)
#pragma unroll
      for (int j = 0; j < 8; ++j) acc[tm][tn][j] = 0.f;

  fetch(0);
  for (int k0 = 0; k0 < K; k0 += BKK) {
    stage();
    __syncthreads();
    if (k0 + BKK < K) fetch(k0 + BKK);   // overlap next-tile loads with WMMA

    // A fragments: K = half*16 + j (contiguous, matching B)
    bf16x8 alo[2], ahi[2];
#pragma unroll
    for (int t = 0; t < 2; ++t) {
      const int row = (wm * 2 + t) * 16 + r;
      alo[t] = *(const bf16x8*)&As[row][half * 16];
      ahi[t] = *(const bf16x8*)&As[row][half * 16 + 8];
    }
    // B fragments: K = half*16 + j, contiguous 32B in transposed tile
    bf16x16 bfrag[2];
#pragma unroll
    for (int t = 0; t < 2; ++t) {
      const int col = (wn * 2 + t) * 16 + r;
      bfrag[t] = *(const bf16x16*)&Bs[col][half * 16];
    }

#pragma unroll
    for (int tm = 0; tm < 2; ++tm) {
      const bf16x16 af = __builtin_shufflevector(
          alo[tm], ahi[tm], 0, 1, 2, 3, 4, 5, 6, 7, 8, 9, 10, 11, 12, 13, 14, 15);
#pragma unroll
      for (int tn = 0; tn < 2; ++tn)
        acc[tm][tn] = __builtin_amdgcn_wmma_f32_16x16x32_bf16(
            false, af, false, bfrag[tn], (short)0, acc[tm][tn], false, false);
    }
    __syncthreads();
  }

  // epilogue (C layout: VGPR j -> row = half*8 + j, col = lane&15)
#pragma unroll
  for (int tm = 0; tm < 2; ++tm) {
    const int mbase = m0 + (wm * 2 + tm) * 16 + half * 8;
#pragma unroll
    for (int tn = 0; tn < 2; ++tn) {
      const int col = n0 + (wn * 2 + tn) * 16 + r;
      if (col < Ncols) {
        const float bv = bias ? bias[col] : 0.f;
#pragma unroll
        for (int j = 0; j < 8; ++j) {
          const int row = mbase + j;
          if (row < Mrows) {
            float v = alpha * acc[tm][tn][j] + bv;
            if (act == 1) v = fmaxf(v, 0.f);
            else if (act == 2) v = tanhf(v) * PI_F;
            C[(size_t)row * ldc + col] = v;
          }
        }
      }
    }
  }
}

// =====================================================================
// Rotation feature builder.
//   proj: [R, Hc*64] rows (R = batch*len), om/th: [R, Hc*Mc]
//   out:  [R, Hc, Mc*64]
// =====================================================================
__global__ void rotate_kernel(const float* __restrict__ proj,
                              const float* __restrict__ om,
                              const float* __restrict__ th,
                              float* __restrict__ out,
                              long long R, int Lq, int isK)
{
  const size_t total = (size_t)R * Hc * Mc * 64;
  for (size_t i = (size_t)blockIdx.x * blockDim.x + threadIdx.x; i < total;
       i += (size_t)gridDim.x * blockDim.x) {
    const int e = (int)(i & 63);
    size_t t = i >> 6;
    const int m = (int)(t & 1); t >>= 1;
    const int h = (int)(t % Hc);
    const size_t row = t / Hc;
    const int l = (int)(row % (size_t)Lq);
    const float pos = (float)l / (float)Lq;
    const float ang = om[row * (Hc * Mc) + h * Mc + m] * pos +
                      th[row * (Hc * Mc) + h * Mc + m];
    float s, c;
    __sincosf(ang, &s, &c);
    const float* q = proj + row * (size_t)(Hc * 64) + h * 64;
    float v;
    if (isK) {
      v = (e < 32) ? q[e] * c - q[e + 32] * s
                   : q[e] * c + q[e - 32] * s;
    } else {
      if (e < 16)      v = q[e] * c - q[e + 16] * s;
      else if (e < 32) v = q[e] * c + q[e - 16] * s;
      else if (e < 48) v = q[e] * c + q[e + 16] * s;
      else             v = q[e] * c - q[e - 16] * s;
    }
    out[row * (size_t)(Hc * 128) + h * 128 + m * 64 + e] = v;
  }
}

// =====================================================================
// Row softmax, in place. One block per row; blockDim.x in {128, 256}.
// =====================================================================
__global__ void softmax_kernel(float* __restrict__ scores, int len)
{
  float* p = scores + (size_t)blockIdx.x * len;
  __shared__ float red[256];
  const int tid = threadIdx.x;
  const int nt = blockDim.x;

  float mx = -3.402823466e38f;
  for (int i = tid; i < len; i += nt) mx = fmaxf(mx, p[i]);
  red[tid] = mx; __syncthreads();
  for (int s = nt >> 1; s > 0; s >>= 1) {
    if (tid < s) red[tid] = fmaxf(red[tid], red[tid + s]);
    __syncthreads();
  }
  mx = red[0]; __syncthreads();

  float sum = 0.f;
  for (int i = tid; i < len; i += nt) {
    float e = __expf(p[i] - mx);
    p[i] = e; sum += e;
  }
  red[tid] = sum; __syncthreads();
  for (int s = nt >> 1; s > 0; s >>= 1) {
    if (tid < s) red[tid] += red[tid + s];
    __syncthreads();
  }
  const float inv = 1.f / red[0];
  for (int i = tid; i < len; i += nt) p[i] *= inv;
}

// =====================================================================
// trend-norm: series_decomp(k=25) + LayerNorm(seasonal) + trend
// x,out: [B, IND, D]; one block per (b,l)
// =====================================================================
__global__ __launch_bounds__(256)
void trend_norm_kernel(const float* __restrict__ x,
                       const float* __restrict__ g, const float* __restrict__ bt,
                       float* __restrict__ out)
{
  const int bl = blockIdx.x;
  const int b = bl / INDc, l = bl % INDc;
  const float* xb = x + (size_t)b * INDc * Dc;
  __shared__ float seas[Dc], trend[Dc];
  __shared__ float red[256];
  __shared__ float s_mu, s_rs;
  const int tid = threadIdx.x;

  float lsum = 0.f, lsq = 0.f;
  for (int d = tid; d < Dc; d += 256) {
    float t = 0.f;
    for (int j = -12; j <= 12; ++j) {
      int ll = l + j;
      ll = ll < 0 ? 0 : (ll > INDc - 1 ? INDc - 1 : ll);
      t += xb[(size_t)ll * Dc + d];
    }
    t *= (1.f / 25.f);
    const float sv = xb[(size_t)l * Dc + d] - t;
    seas[d] = sv; trend[d] = t;
    lsum += sv; lsq += sv * sv;
  }
  red[tid] = lsum; __syncthreads();
  for (int s = 128; s > 0; s >>= 1) { if (tid < s) red[tid] += red[tid + s]; __syncthreads(); }
  const float tot = red[0]; __syncthreads();
  red[tid] = lsq; __syncthreads();
  for (int s = 128; s > 0; s >>= 1) { if (tid < s) red[tid] += red[tid + s]; __syncthreads(); }
  if (tid == 0) {
    const float mu = tot / (float)Dc;
    const float var = red[0] / (float)Dc - mu * mu;
    s_mu = mu;
    s_rs = rsqrtf(var + 1e-5f);
  }
  __syncthreads();
  const float mu = s_mu, rs = s_rs;
  for (int d = tid; d < Dc; d += 256)
    out[(size_t)bl * Dc + d] = (seas[d] - mu) * rs * g[d] + bt[d] + trend[d];
}

// =====================================================================
// Penalty reductions
// =====================================================================
__global__ void pen_init_kernel(float* __restrict__ tgt)
{
  if (blockIdx.x == 0 && threadIdx.x == 0) { tgt[0] = 0.f; tgt[1] = 0.f; }
}

__global__ __launch_bounds__(256)
void pen_sq_kernel(const float* __restrict__ x, long long n, float scale,
                   float* __restrict__ tgt)
{
  float s = 0.f;
  for (size_t i = (size_t)blockIdx.x * blockDim.x + threadIdx.x; i < (size_t)n;
       i += (size_t)gridDim.x * blockDim.x)
    s += x[i] * x[i];
  __shared__ float red[256];
  red[threadIdx.x] = s; __syncthreads();
  for (int t = 128; t > 0; t >>= 1) {
    if (threadIdx.x < t) red[threadIdx.x] += red[threadIdx.x + t];
    __syncthreads();
  }
  if (threadIdx.x == 0) atomicAdd(tgt, scale * red[0]);
}

// sum over (r, l<Lq-1, c) of (x[r][l+1][c]-x[r][l][c])^2 ; x: [R, Lq, Cc]
__global__ __launch_bounds__(256)
void pen_diff_kernel(const float* __restrict__ x, long long R, int Lq, int Cc,
                     float scale, float* __restrict__ tgt)
{
  const size_t n = (size_t)R * (Lq - 1) * Cc;
  float s = 0.f;
  for (size_t i = (size_t)blockIdx.x * blockDim.x + threadIdx.x; i < n;
       i += (size_t)gridDim.x * blockDim.x) {
    const int c = (int)(i % Cc);
    size_t t = i / Cc;
    const int l = (int)(t % (size_t)(Lq - 1));
    const size_t r = t / (size_t)(Lq - 1);
    const float* p = x + (r * (size_t)Lq + l) * Cc + c;
    const float d = p[Cc] - p[0];
    s += d * d;
  }
  __shared__ float red[256];
  red[threadIdx.x] = s; __syncthreads();
  for (int t = 128; t > 0; t >>= 1) {
    if (threadIdx.x < t) red[threadIdx.x] += red[threadIdx.x + t];
    __syncthreads();
  }
  if (threadIdx.x == 0) atomicAdd(tgt, scale * red[0]);
}

// =====================================================================
// host side
// =====================================================================
static inline void launch_gemm(const float* A, const float* W, const float* bias,
                               float* C, int M, int N, int K,
                               int lda, int ldb, int ldc,
                               long long aB, long long aH, long long bB, long long bH,
                               long long cB, long long cH,
                               int Z, int Hcnt, int transB, float alpha, int act,
                               hipStream_t s)
{
  dim3 grid((M + BM - 1) / BM, (N + BN - 1) / BN, Z);
  wmma_gemm_bf16<<<grid, 256, 0, s>>>(A, W, bias, C, M, N, K, lda, ldb, ldc,
                                      aB, aH, bB, bH, cB, cH, Hcnt, transB, alpha, act);
}

extern "C" void kernel_launch(void* const* d_in, const int* in_sizes, int n_in,
                              void* d_out, int out_size, void* d_ws, size_t ws_size,
                              hipStream_t stream)
{
  (void)in_sizes; (void)n_in; (void)out_size; (void)ws_size;

  const float* queries = (const float*)d_in[0];
  const float* keys    = (const float*)d_in[1];
  const float* values  = (const float*)d_in[2];
  const float* Wq  = (const float*)d_in[3];  const float* bq  = (const float*)d_in[4];
  const float* Wk  = (const float*)d_in[5];  const float* bk  = (const float*)d_in[6];
  const float* Wv  = (const float*)d_in[7];  const float* bv  = (const float*)d_in[8];
  const float* Wqo = (const float*)d_in[9];  const float* bqo = (const float*)d_in[10];
  const float* Wko = (const float*)d_in[11]; const float* bko = (const float*)d_in[12];
  const float* Wqt = (const float*)d_in[13]; const float* bqt = (const float*)d_in[14];
  const float* Wkt = (const float*)d_in[15]; const float* bkt = (const float*)d_in[16];
  const float* Wo  = (const float*)d_in[17]; const float* bo  = (const float*)d_in[18];
  const float* Imat = (const float*)d_in[19];
  const float* ln_g = (const float*)d_in[20];
  const float* ln_b = (const float*)d_in[21];

  float* out = (float*)d_out;
  float* pen = out + (size_t)Bc * Lc * Dc;   // [om_pen, th_pen]
  float* ws  = (float*)d_ws;

  const int HM = Hc * Mc;                    // 16
  const long long BS = (long long)Bc * Sc;   // 24576
  const long long BL = (long long)Bc * Lc;   // 24576
  const long long BI = (long long)Bc * INDc; // 768

  // ---- workspace layout (floats); layer1 reuses layer0's big buffers ----
  const size_t o_kproj0   = 0;                                       // BS*D (reused: qproj1)
  const size_t o_vproj0   = o_kproj0 + (size_t)BS * Dc;              // BS*D (reused: attnout1)
  const size_t o_Kr0      = o_vproj0 + (size_t)BS * Dc;              // BS*H*128 (reused: Qr1)
  const size_t o_scores0  = o_Kr0 + (size_t)BS * Hc * 128;           // B*H*IND*S (reused: scores1)
  const size_t o_kom0     = o_scores0 + (size_t)Bc * Hc * INDc * Sc; // BS*16 (reused: q_om1)
  const size_t o_kth0     = o_kom0 + (size_t)BS * HM;                // BS*16 (reused: q_th1)
  const size_t o_qproj0   = o_kth0 + (size_t)BS * HM;                // IND*D
  const size_t o_qom0     = o_qproj0 + (size_t)INDc * Dc;            // IND*16
  const size_t o_qth0     = o_qom0 + (size_t)INDc * HM;              // IND*16
  const size_t o_Qr0      = o_qth0 + (size_t)INDc * HM;              // IND*H*128
  const size_t o_attn0    = o_Qr0 + (size_t)INDc * Hc * 128;         // B*IND*D
  const size_t o_ind0     = o_attn0 + (size_t)BI * Dc;               // B*IND*D
  const size_t o_ind1     = o_ind0 + (size_t)BI * Dc;                // B*IND*D
  const size_t o_kproj1   = o_ind1 + (size_t)BI * Dc;                // BI*D
  const size_t o_vproj1   = o_kproj1 + (size_t)BI * Dc;              // BI*D
  const size_t o_kom1     = o_vproj1 + (size_t)BI * Dc;              // BI*16
  const size_t o_kth1     = o_kom1 + (size_t)BI * HM;                // BI*16
  const size_t o_Kr1      = o_kth1 + (size_t)BI * HM;                // BI*H*128

  // aliases for reused regions
  const size_t o_qproj1  = o_kproj0;
  const size_t o_attn1   = o_vproj0;
  const size_t o_Qr1     = o_Kr0;
  const size_t o_scores1 = o_scores0;
  const size_t o_qom1    = o_kom0;
  const size_t o_qth1    = o_kth0;

  const size_t DD = (size_t)Dc * Dc;
  const float inv16 = 1.f / 16.f;   // (1/M) * (1/sqrt(E)) = (1/2)*(1/8)

  pen_init_kernel<<<1, 32, 0, stream>>>(pen);

  // =============== layer 0: inducings attend to keys/values ===============
  launch_gemm(keys,   Wk,        bk,        ws + o_kproj0, (int)BS, Dc, Dc, Dc, Dc, Dc,
              0,0,0,0,0,0, 1,1, 0, 1.f, 0, stream);
  launch_gemm(values, Wv,        bv,        ws + o_vproj0, (int)BS, Dc, Dc, Dc, Dc, Dc,
              0,0,0,0,0,0, 1,1, 0, 1.f, 0, stream);
  launch_gemm(keys,   Wko,       bko,       ws + o_kom0,   (int)BS, HM, Dc, Dc, HM, HM,
              0,0,0,0,0,0, 1,1, 0, 1.f, 1, stream);
  launch_gemm(keys,   Wkt,       bkt,       ws + o_kth0,   (int)BS, HM, Dc, Dc, HM, HM,
              0,0,0,0,0,0, 1,1, 0, 1.f, 2, stream);
  // q-side on inducings (identical across batch -> computed once)
  launch_gemm(Imat,   Wq,        bq,        ws + o_qproj0, INDc, Dc, Dc, Dc, Dc, Dc,
              0,0,0,0,0,0, 1,1, 0, 1.f, 0, stream);
  launch_gemm(Imat,   Wqo,       bqo,       ws + o_qom0,   INDc, HM, Dc, Dc, HM, HM,
              0,0,0,0,0,0, 1,1, 0, 1.f, 1, stream);
  launch_gemm(Imat,   Wqt,       bqt,       ws + o_qth0,   INDc, HM, Dc, Dc, HM, HM,
              0,0,0,0,0,0, 1,1, 0, 1.f, 2, stream);

  {
    size_t tot = (size_t)INDc * Hc * Mc * 64;
    rotate_kernel<<<(int)((tot + 255) / 256), 256, 0, stream>>>(
        ws + o_qproj0, ws + o_qom0, ws + o_qth0, ws + o_Qr0, INDc, INDc, 0);
    tot = (size_t)BS * Hc * Mc * 64;
    rotate_kernel<<<(int)((tot + 255) / 256), 256, 0, stream>>>(
        ws + o_kproj0, ws + o_kom0, ws + o_kth0, ws + o_Kr0, BS, Sc, 1);
  }

  // scores0[b,h,l,s] = (Qr0 @ Kr0^T)/16  : M=IND, N=S, K=128
  launch_gemm(ws + o_Qr0, ws + o_Kr0, nullptr, ws + o_scores0,
              INDc, Sc, 128, Hc * 128, Hc * 128, Sc,
              /*aB*/0, /*aH*/128,
              /*bB*/(long long)Sc * Hc * 128, /*bH*/128,
              /*cB*/(long long)Hc * INDc * Sc, /*cH*/(long long)INDc * Sc,
              Bc * Hc, Hc, /*transB*/1, inv16, 0, stream);

  softmax_kernel<<<Bc * Hc * INDc, 256, 0, stream>>>(ws + o_scores0, Sc);

  // attn0[b,l,h,e] = A @ V : M=IND, N=64, K=S
  launch_gemm(ws + o_scores0, ws + o_vproj0, nullptr, ws + o_attn0,
              INDc, Ec, Sc, Sc, Dc, Dc,
              (long long)Hc * INDc * Sc, (long long)INDc * Sc,
              (long long)Sc * Dc, 64,
              (long long)INDc * Dc, 64,
              Bc * Hc, Hc, 0, 1.f, 0, stream);

  // out proj
  launch_gemm(ws + o_attn0, Wo, bo, ws + o_ind0, (int)BI, Dc, Dc, Dc, Dc, Dc,
              0,0,0,0,0,0, 1,1, 0, 1.f, 0, stream);

  trend_norm_kernel<<<(int)BI, 256, 0, stream>>>(ws + o_ind0, ln_g, ln_b, ws + o_ind1);

  // layer0 penalties (q-side broadcast over batch -> scale B)
  pen_diff_kernel<<<512, 256, 0, stream>>>(ws + o_qom0, 1, INDc, HM, (float)Bc, pen + 0);
  pen_diff_kernel<<<1024, 256, 0, stream>>>(ws + o_kom0, Bc, Sc, HM, 1.f, pen + 0);
  pen_sq_kernel<<<512, 256, 0, stream>>>(ws + o_qth0, (long long)INDc * HM, (float)Bc, pen + 1);
  pen_sq_kernel<<<1024, 256, 0, stream>>>(ws + o_kth0, BS * HM, 1.f, pen + 1);

  // =============== layer 1: queries attend to inducings ===============
  launch_gemm(queries, Wq + DD,  bq + Dc,   ws + o_qproj1, (int)BL, Dc, Dc, Dc, Dc, Dc,
              0,0,0,0,0,0, 1,1, 0, 1.f, 0, stream);
  launch_gemm(queries, Wqo + (size_t)Dc * HM, bqo + HM, ws + o_qom1, (int)BL, HM, Dc, Dc, HM, HM,
              0,0,0,0,0,0, 1,1, 0, 1.f, 1, stream);
  launch_gemm(queries, Wqt + (size_t)Dc * HM, bqt + HM, ws + o_qth1, (int)BL, HM, Dc, Dc, HM, HM,
              0,0,0,0,0,0, 1,1, 0, 1.f, 2, stream);
  launch_gemm(ws + o_ind1, Wk + DD, bk + Dc, ws + o_kproj1, (int)BI, Dc, Dc, Dc, Dc, Dc,
              0,0,0,0,0,0, 1,1, 0, 1.f, 0, stream);
  launch_gemm(ws + o_ind1, Wv + DD, bv + Dc, ws + o_vproj1, (int)BI, Dc, Dc, Dc, Dc, Dc,
              0,0,0,0,0,0, 1,1, 0, 1.f, 0, stream);
  launch_gemm(ws + o_ind1, Wko + (size_t)Dc * HM, bko + HM, ws + o_kom1, (int)BI, HM, Dc, Dc, HM, HM,
              0,0,0,0,0,0, 1,1, 0, 1.f, 1, stream);
  launch_gemm(ws + o_ind1, Wkt + (size_t)Dc * HM, bkt + HM, ws + o_kth1, (int)BI, HM, Dc, Dc, HM, HM,
              0,0,0,0,0,0, 1,1, 0, 1.f, 2, stream);

  {
    size_t tot = (size_t)BL * Hc * Mc * 64;
    rotate_kernel<<<(int)((tot + 255) / 256), 256, 0, stream>>>(
        ws + o_qproj1, ws + o_qom1, ws + o_qth1, ws + o_Qr1, BL, Lc, 0);
    tot = (size_t)BI * Hc * Mc * 64;
    rotate_kernel<<<(int)((tot + 255) / 256), 256, 0, stream>>>(
        ws + o_kproj1, ws + o_kom1, ws + o_kth1, ws + o_Kr1, BI, INDc, 1);
  }

  // scores1: M=L, N=IND, K=128
  launch_gemm(ws + o_Qr1, ws + o_Kr1, nullptr, ws + o_scores1,
              Lc, INDc, 128, Hc * 128, Hc * 128, INDc,
              (long long)Lc * Hc * 128, 128,
              (long long)INDc * Hc * 128, 128,
              (long long)Hc * Lc * INDc, (long long)Lc * INDc,
              Bc * Hc, Hc, 1, inv16, 0, stream);

  softmax_kernel<<<Bc * Hc * Lc, 128, 0, stream>>>(ws + o_scores1, INDc);

  // attn1: M=L, N=64, K=IND
  launch_gemm(ws + o_scores1, ws + o_vproj1, nullptr, ws + o_attn1,
              Lc, Ec, INDc, INDc, Dc, Dc,
              (long long)Hc * Lc * INDc, (long long)Lc * INDc,
              (long long)INDc * Dc, 64,
              (long long)Lc * Dc, 64,
              Bc * Hc, Hc, 0, 1.f, 0, stream);

  // final output projection -> d_out
  launch_gemm(ws + o_attn1, Wo + DD, bo + Dc, out, (int)BL, Dc, Dc, Dc, Dc, Dc,
              0,0,0,0,0,0, 1,1, 0, 1.f, 0, stream);

  // layer1 penalties
  pen_diff_kernel<<<1024, 256, 0, stream>>>(ws + o_qom1, Bc, Lc, HM, 1.f, pen + 0);
  pen_diff_kernel<<<512, 256, 0, stream>>>(ws + o_kom1, Bc, INDc, HM, 1.f, pen + 0);
  pen_sq_kernel<<<1024, 256, 0, stream>>>(ws + o_qth1, BL * HM, 1.f, pen + 1);
  pen_sq_kernel<<<512, 256, 0, stream>>>(ws + o_kth1, BI * HM, 1.f, pen + 1);
}
